// GruTl_21466246545968
// MI455X (gfx1250) — matrix operand order repaired and checked by reading, not compiled, not run on hardware
//
#include <hip/hip_runtime.h>

typedef __attribute__((ext_vector_type(16))) _Float16 v16h;
typedef __attribute__((ext_vector_type(8)))  _Float16 v8h;
typedef __attribute__((ext_vector_type(8)))  float    v8f;

#define NTHREADS 256
#define NWAVES   8

__device__ __forceinline__ v8f wmma_f16(v16h a, v16h b, v8f c) {
  // D = A(16x32 f16) * B(32x16 f16) + C(16x16 f32)
  return __builtin_amdgcn_wmma_f32_16x16x32_f16(false, a, false, b, (short)0, c,
                                                false, false);
}

// A fragment (16x32 f16) from row-major staging: row stride 32 halves.
// Lane(m,hi): K-sequence is [8hi..8hi+7] ++ [16+8hi..16+8hi+7] -> two b128 loads.
__device__ __forceinline__ v16h load_a(const _Float16* base, int lane) {
  const int m = lane & 15, hi = lane >> 4;
  const v8h a0 = *(const v8h*)(base + m * 32 + 8 * hi);
  const v8h a1 = *(const v8h*)(base + m * 32 + 16 + 8 * hi);
  return __builtin_shufflevector(a0, a1, 0, 1, 2, 3, 4, 5, 6, 7,
                                 8, 9, 10, 11, 12, 13, 14, 15);
}

// B fragment (32x16 f16) from column-major K-contiguous staging:
// addr = col*32 + k. Lane(n,hi): K-sequence [8hi..8hi+7] ++ [16+8hi..16+8hi+7] -> two b128 loads.
__device__ __forceinline__ v16h load_b(const _Float16* base, int lane) {
  const int n = lane & 15, hi = lane >> 4;
  const v8h b0 = *(const v8h*)(base + n * 32 + 8 * hi);
  const v8h b1 = *(const v8h*)(base + n * 32 + 16 + 8 * hi);
  return __builtin_shufflevector(b0, b1, 0, 1, 2, 3, 4, 5, 6, 7,
                                 8, 9, 10, 11, 12, 13, 14, 15);
}

// Stages A+B of the 3-mode contraction for all 3 gates (fully unrolled).
// INb : f16 col-major (bc)*32 + a          S1h : f16 (d*8+c)*32 + b (b>=16 zero)
// S2h : f16 (d*16+e)*32 + c (c>=8 zero)
// Contains one internal barrier (A->B). Caller barriers around it.
__device__ __forceinline__ void tclAB(const _Float16* INb, const _Float16* A1,
                                      const _Float16* A2, _Float16* S1h,
                                      _Float16* S2h, int wave, int lane) {
  const int n = lane & 15, hi = lane >> 4;
  // ---- Stage A: S1[g][d,(b,c)] = M1 (32x32) @ IN (32x128); nt = wave ----
  {
    const v16h b = load_b(INb + wave * 512, lane);        // shared by all 6 tiles
    const int bc = wave * 16 + n, bb = bc >> 3, cc = bc & 7;
#pragma unroll
    for (int g = 0; g < 3; ++g) {
#pragma unroll
      for (int mt = 0; mt < 2; ++mt) {
        v16h a = load_a(A1 + g * 1024 + mt * 512, lane);
        v8f c = {};
        c = wmma_f16(a, b, c);
        _Float16* O = S1h + g * 8192 + cc * 32 + bb;
#pragma unroll
        for (int j = 0; j < 8; ++j) {
          const int d = mt * 16 + 8 * hi + j;
          O[d * 256] = (_Float16)c[j];                    // (d*8+cc)*32 + bb
        }
      }
    }
  }
  __syncthreads();
  // ---- Stage B: S2[g][e,(d,c)] = M2 (16x16 padK32) @ S1 (16x256) ----
#pragma unroll
  for (int g = 0; g < 3; ++g) {
    const v16h a = load_a(A2 + g * 512, lane);            // shared by 2 tiles
#pragma unroll
    for (int half = 0; half < 2; ++half) {
      const int nt = wave + 8 * half;
      v16h b = load_b(S1h + g * 8192 + nt * 512, lane);
      v8f c = {};
      c = wmma_f16(a, b, c);
      const int d = 2 * nt + (n >> 3), cc = n & 7;        // col = nt*16+n
      _Float16* O = S2h + g * 16384 + d * 512 + cc;
#pragma unroll
      for (int j = 0; j < 8; ++j) {
        const int e = 8 * hi + j;
        O[e * 32] = (_Float16)c[j];                       // (d*16+e)*32 + cc
      }
    }
  }
}

__global__ void __launch_bounds__(NTHREADS, 1)
grutl_kernel(const float* __restrict__ x, const float* __restrict__ W1,
             const float* __restrict__ W2, const float* __restrict__ W3,
             const float* __restrict__ U1, const float* __restrict__ U2,
             const float* __restrict__ U3, float* __restrict__ out) {
  extern __shared__ float sm[];
  float* lH = sm;           // 4096   f32 hidden state (d*128+e*8+f)
  float* XW = sm + 4096;    // 3*4096 f32 x-projection
  _Float16* fh = (_Float16*)(sm + 16384);  // f16 arena (94208 halves)
  _Float16* lXh  = fh;             // 4096  x staged (bc*32 + a)
  _Float16* lHh  = fh + 4096;      // 4096  h staged (bc*32 + a)
  _Float16* S1h  = fh + 8192;      // 24576
  _Float16* S2h  = fh + 32768;     // 49152
  _Float16* wA1W = fh + 81920;     // 3072
  _Float16* wA2W = fh + 84992;     // 1536
  _Float16* wA3W = fh + 86528;     // 1536
  _Float16* wA1U = fh + 88064;     // 3072
  _Float16* wA2U = fh + 91136;     // 1536
  _Float16* wA3U = fh + 92672;     // 1536 -> total 94208 halves

  const int tid = threadIdx.x;
  const int lane = tid & 31, wave = tid >> 5;
  const int nb = blockIdx.x;
  const int n = lane & 15, hi = lane >> 4;

  // Zero the f16 arena once (establishes all K/M zero-padding), zero h0.
  {
    v8h z = {};
    v8h* p = (v8h*)fh;
    for (int i = tid; i < 94208 / 8; i += NTHREADS) p[i] = z;
    for (int i = tid; i < 4096; i += NTHREADS) lH[i] = 0.0f;
  }
  __syncthreads();
  // Stage weights to f16 fragment layouts (row-major, 32-half rows, zero-padded).
  for (int i = tid; i < 3072; i += NTHREADS) {
    wA1W[i] = (_Float16)W1[i];
    wA1U[i] = (_Float16)U1[i];
  }
  for (int i = tid; i < 768; i += NTHREADS) {
    const int g = i >> 8, r = i & 255, e = r >> 4, b = r & 15;
    wA2W[g * 512 + e * 32 + b] = (_Float16)W2[i];
    wA2U[g * 512 + e * 32 + b] = (_Float16)U2[i];
  }
  for (int i = tid; i < 192; i += NTHREADS) {
    const int g = i >> 6, r = i & 63, f = r >> 3, c = r & 7;
    wA3W[g * 512 + f * 32 + c] = (_Float16)W3[i];
    wA3U[g * 512 + f * 32 + c] = (_Float16)U3[i];
  }
  __syncthreads();

  float* hlast = out + (size_t)64 * 128 * 4096;

  for (int l = 0; l < 128; ++l) {
    // Stream x(nb,l) coalesced, transpose into f16 staging (bc*32 + a).
    const float* gx = x + ((size_t)nb * 128 + l) * 4096;
    if (l + 1 < 128) {  // prefetch next slab into cache (global_prefetch_b8)
      __builtin_prefetch(gx + 4096 + tid * 16, 0, 1);
    }
    for (int i = tid; i < 4096; i += NTHREADS) {
      lXh[(i & 127) * 32 + (i >> 7)] = (_Float16)gx[i];
    }
    __syncthreads();

    // ---- W path: XW[g][d,e,f] = TCL(x) ----
    tclAB(lXh, wA1W, wA2W, S1h, S2h, wave, lane);
    __syncthreads();
#pragma unroll
    for (int g = 0; g < 3; ++g) {           // stage C -> XW (f32, contiguous)
      const v16h a = load_a(wA3W + g * 512, lane);
#pragma unroll
      for (int q = 0; q < 4; ++q) {
        const int nt = wave + 8 * q;        // d = nt, e = n
        v16h b = load_b(S2h + g * 16384 + nt * 512, lane);
        v8f c = {};
        c = wmma_f16(a, b, c);
        if (hi == 0) {                      // rows f=0..7 valid -> 2x b128
          float* O = XW + g * 4096 + nt * 128 + n * 8;
          const float4 lo4 = {c[0], c[1], c[2], c[3]};
          const float4 hi4 = {c[4], c[5], c[6], c[7]};
          *(float4*)(O) = lo4;
          *(float4*)(O + 4) = hi4;
        }
      }
    }
    // No barrier needed: U stage A touches only S1h (disjoint from XW/S2h use).
    // ---- U path stages A+B ----
    tclAB(lHh, wA1U, wA2U, S1h, S2h, wave, lane);
    __syncthreads();
    // ---- Fused U stage C + GRU gates (register-resident) ----
    float* o = out + ((size_t)nb * 128 + l) * 4096;
#pragma unroll
    for (int q = 0; q < 4; ++q) {
      const int nt = wave + 8 * q;          // d = nt, e = n
      v8f cU[3];
#pragma unroll
      for (int g = 0; g < 3; ++g) {
        const v16h a = load_a(wA3U + g * 512, lane);
        v16h b = load_b(S2h + g * 16384 + nt * 512, lane);
        v8f c = {};
        cU[g] = wmma_f16(a, b, c);
      }
      if (hi == 0) {
        const int base = nt * 128 + n * 8;  // d*128 + e*8 (+f)
        float xw0[8], xw1[8], xw2[8], hp[8], hn[8];
        *(float4*)&xw0[0] = *(const float4*)(XW + base);
        *(float4*)&xw0[4] = *(const float4*)(XW + base + 4);
        *(float4*)&xw1[0] = *(const float4*)(XW + 4096 + base);
        *(float4*)&xw1[4] = *(const float4*)(XW + 4096 + base + 4);
        *(float4*)&xw2[0] = *(const float4*)(XW + 8192 + base);
        *(float4*)&xw2[4] = *(const float4*)(XW + 8192 + base + 4);
        *(float4*)&hp[0]  = *(const float4*)(lH + base);
        *(float4*)&hp[4]  = *(const float4*)(lH + base + 4);
#pragma unroll
        for (int f = 0; f < 8; ++f) {
          const float z  = 1.0f / (1.0f + __expf(-(xw0[f] + cU[0][f])));
          const float r  = 1.0f / (1.0f + __expf(-(xw1[f] + cU[1][f])));
          const float hh = tanhf(xw2[f] + r * cU[2][f]);
          hn[f] = z * hp[f] + (1.0f - z) * hh;
        }
        *(float4*)(lH + base)     = *(float4*)&hn[0];
        *(float4*)(lH + base + 4) = *(float4*)&hn[4];
        *(float4*)(o + base)      = *(float4*)&hn[0];   // 512B contig / 16 lanes
        *(float4*)(o + base + 4)  = *(float4*)&hn[4];
#pragma unroll
        for (int f = 0; f < 8; ++f) {                   // f16 staged copy of h
          lHh[(n * 8 + f) * 32 + nt] = (_Float16)hn[f]; // (bc)*32 + a
        }
      }
    }
    __syncthreads();  // lH/lHh/S2h stable before next step's staging phases
  }

  for (int i = tid; i < 4096; i += NTHREADS) {
    hlast[(size_t)nb * 4096 + i] = lH[i];
  }
}

extern "C" void kernel_launch(void* const* d_in, const int* in_sizes, int n_in,
                              void* d_out, int out_size, void* d_ws, size_t ws_size,
                              hipStream_t stream) {
  (void)in_sizes; (void)n_in; (void)out_size; (void)d_ws; (void)ws_size;
  const float* x  = (const float*)d_in[0];
  const float* W1 = (const float*)d_in[1];
  const float* W2 = (const float*)d_in[2];
  const float* W3 = (const float*)d_in[3];
  const float* U1 = (const float*)d_in[4];
  const float* U2 = (const float*)d_in[5];
  const float* U3 = (const float*)d_in[6];
  float* out = (float*)d_out;

  const size_t smem = 16384 * sizeof(float) + 94208 * sizeof(_Float16);  // 253952 B
  hipFuncSetAttribute((const void*)grutl_kernel,
                      hipFuncAttributeMaxDynamicSharedMemorySize, (int)smem);
  grutl_kernel<<<64, NTHREADS, smem, stream>>>(x, W1, W2, W3, U1, U2, U3, out);
}
